// PretrainedGNNEncoder_532575945203
// MI455X (gfx1250) — matrix operand order repaired and checked by reading, not compiled, not run on hardware
//
#include <hip/hip_runtime.h>
#include <hip/hip_bf16.h>

#define NN 100000      // nodes
#define NE 1600000     // edges
#define INC 128        // in channels
#define DD 64          // hidden dim
#define EDC 16         // edge dim
#define NL 6           // layers

typedef __attribute__((ext_vector_type(16))) _Float16 v16h;
typedef __attribute__((ext_vector_type(8)))  float    v8f;

// ---------------------------------------------------------------- utilities

__global__ void cvt_f32_to_f16_kernel(const float* __restrict__ in,
                                      _Float16* __restrict__ out, int n) {
    int i = blockIdx.x * blockDim.x + threadIdx.x;
    if (i < n) out[i] = (_Float16)in[i];
}

__global__ void zero_u32_kernel(unsigned int* __restrict__ p, int n) {
    int i = blockIdx.x * blockDim.x + threadIdx.x;
    if (i < n) p[i] = 0u;
}

// -------------------------------------------- fused QKVS WMMA GEMM
// Computes four projections of the same X in one pass:
//   Q = X*Wq+bq, K = X*Wk+bk, V = X*Wv+bv, S = X*Ws+bs
// X: [M,Kdim] f16 row-major; W*: [Kdim,64] f16; b*: [64] f32; outputs [M,64] f32.
// One wave per 16x16 output tile; A fragment loaded once, 4 WMMAs per K-step.
__global__ __launch_bounds__(256)
void gemm_qkvs_wmma_kernel(const _Float16* __restrict__ X,
                           const _Float16* __restrict__ Wq,
                           const _Float16* __restrict__ Wk,
                           const _Float16* __restrict__ Wv,
                           const _Float16* __restrict__ Ws,
                           const float* __restrict__ bq,
                           const float* __restrict__ bk,
                           const float* __restrict__ bv,
                           const float* __restrict__ bs,
                           float* __restrict__ Q, float* __restrict__ Ko,
                           float* __restrict__ V, float* __restrict__ S,
                           int M, int Kdim) {
    const int wavesPerBlock = blockDim.x >> 5;
    const int wave  = blockIdx.x * wavesPerBlock + (threadIdx.x >> 5);
    const int tiles = (M >> 4) * (DD >> 4);
    if (wave >= tiles) return;               // wave-uniform: EXEC stays all-ones

    const int tr   = wave >> 2;              // row tile (DD/16 == 4 col tiles)
    const int tc   = wave & 3;
    const int lane = threadIdx.x & 31;
    const int half = lane >> 4;              // 0: lanes 0-15, 1: lanes 16-31
    const int l15  = lane & 15;
    const int arow = (tr << 4) + l15;        // A: lane -> M row
    const int n0   = tc << 4;

    v8f accQ = {}, accK = {}, accV = {}, accS = {};
    for (int kb = 0; kb < Kdim; kb += 32) {
        v16h a, bQ, bK, bV, bS;
        // A 16x32 f16 layout: lane l (m=l&15,h=l>>4), VGPR v holds K pair p,p+1
        //   p = 2v+8h (v<4), 16+2(v-4)+8h (v>=4)
        const _Float16* xrow = X + arow * Kdim + kb;
        // B 32x16 f16 layout: lane -> N column (l15), element pair -> K pair p,p+1 (same as A)
        const int wofs = kb * DD + n0 + l15;
#pragma unroll
        for (int v = 0; v < 8; ++v) {
            const int p = (v < 4) ? (2 * v + 8 * half) : (16 + 2 * (v - 4) + 8 * half);
            a[2 * v]      = xrow[p];
            a[2 * v + 1]  = xrow[p + 1];
            bQ[2 * v]     = Wq[wofs + p * DD];
            bQ[2 * v + 1] = Wq[wofs + (p + 1) * DD];
            bK[2 * v]     = Wk[wofs + p * DD];
            bK[2 * v + 1] = Wk[wofs + (p + 1) * DD];
            bV[2 * v]     = Wv[wofs + p * DD];
            bV[2 * v + 1] = Wv[wofs + (p + 1) * DD];
            bS[2 * v]     = Ws[wofs + p * DD];
            bS[2 * v + 1] = Ws[wofs + (p + 1) * DD];
        }
        accQ = __builtin_amdgcn_wmma_f32_16x16x32_f16(
                   false, a, false, bQ, (short)0, accQ, false, false);
        accK = __builtin_amdgcn_wmma_f32_16x16x32_f16(
                   false, a, false, bK, (short)0, accK, false, false);
        accV = __builtin_amdgcn_wmma_f32_16x16x32_f16(
                   false, a, false, bV, (short)0, accV, false, false);
        accS = __builtin_amdgcn_wmma_f32_16x16x32_f16(
                   false, a, false, bS, (short)0, accS, false, false);
    }
    // C/D layout: lane -> N (l15), VGPR r -> M = r + 8*half
    const int ccol = n0 + l15;
    const float vbq = bq[ccol], vbk = bk[ccol], vbv = bv[ccol], vbs = bs[ccol];
#pragma unroll
    for (int r = 0; r < 8; ++r) {
        const int m = (tr << 4) + r + (half << 3);
        Q[m * DD + ccol]  = accQ[r] + vbq;
        Ko[m * DD + ccol] = accK[r] + vbk;
        V[m * DD + ccol]  = accV[r] + vbv;
        S[m * DD + ccol]  = accS[r] + vbs;
    }
}

// ------------------------------------------------------------- edge kernels
// e_c (edge lift) is recomputed from LDS-cached We/be instead of materializing
// a 410MB E x 64 buffer: 1 KFLOP/edge of VALU vs ~768B/edge of L2 traffic.

__device__ __forceinline__ unsigned int ordered_bits(float x) {
    unsigned int u = __float_as_uint(x);
    return (u & 0x80000000u) ? ~u : (u | 0x80000000u);
}

__global__ __launch_bounds__(256)
void edge_logits_kernel(const int* __restrict__ src, const int* __restrict__ dst,
                        const float* __restrict__ ea,
                        const float* __restrict__ We, const float* __restrict__ be,
                        const float* __restrict__ q,  const float* __restrict__ k,
                        float* __restrict__ logits, unsigned int* __restrict__ amax) {
    __shared__ float sWe[EDC * DD];
    __shared__ float sbe[DD];
    for (int i = threadIdx.x; i < EDC * DD; i += blockDim.x) sWe[i] = We[i];
    if (threadIdx.x < DD) sbe[threadIdx.x] = be[threadIdx.x];
    __syncthreads();

    const int ed = blockIdx.x * blockDim.x + threadIdx.x;
    if (ed >= NE) return;
    const int s = src[ed], d = dst[ed];
    float eav[EDC];
#pragma unroll
    for (int j = 0; j < EDC; ++j) eav[j] = ea[ed * EDC + j];
    const float* qr = q + d * DD;
    const float* kr = k + s * DD;
    float acc = 0.f;
    for (int c = 0; c < DD; ++c) {
        float e = sbe[c];
#pragma unroll
        for (int j = 0; j < EDC; ++j) e = fmaf(eav[j], sWe[j * DD + c], e);
        acc = fmaf(qr[c], kr[c] + e, acc);
    }
    acc *= 0.125f;                      // 1/sqrt(64)
    logits[ed] = acc;
    atomicMax(&amax[d], ordered_bits(acc));
}

__global__ void amax_finalize_kernel(unsigned int* __restrict__ amax, int n) {
    const int i = blockIdx.x * blockDim.x + threadIdx.x;
    if (i >= n) return;
    const unsigned int u = amax[i];
    float x = 0.f;                      // empty segment (-inf) -> 0, like ref
    if (u != 0u) {
        const unsigned int bits = (u & 0x80000000u) ? (u & 0x7fffffffu) : ~u;
        x = __uint_as_float(bits);
        if (!isfinite(x)) x = 0.f;
    }
    ((float*)amax)[i] = x;              // reinterpret in place as float
}

__global__ __launch_bounds__(256)
void edge_exp_kernel(const int* __restrict__ dst,
                     float* __restrict__ logits,          // in: logits, out: w
                     const float* __restrict__ amax,
                     float* __restrict__ den) {
    const int ed = blockIdx.x * blockDim.x + threadIdx.x;
    if (ed >= NE) return;
    const int d = dst[ed];
    const float w = expf(logits[ed] - amax[d]);
    logits[ed] = w;
    atomicAdd(&den[d], w);
}

__global__ __launch_bounds__(256)
void edge_agg_kernel(const int* __restrict__ src, const int* __restrict__ dst,
                     const float* __restrict__ ea,
                     const float* __restrict__ We, const float* __restrict__ be,
                     const float* __restrict__ v,  const float* __restrict__ wexp,
                     const float* __restrict__ den, float* __restrict__ agg) {
    __shared__ float sWe[EDC * DD];
    __shared__ float sbe[DD];
    for (int i = threadIdx.x; i < EDC * DD; i += blockDim.x) sWe[i] = We[i];
    if (threadIdx.x < DD) sbe[threadIdx.x] = be[threadIdx.x];
    __syncthreads();

    const int ed = blockIdx.x * blockDim.x + threadIdx.x;
    if (ed >= NE) return;
    const int s = src[ed], d = dst[ed];
    const float coef = wexp[ed] / (den[d] + 1e-16f);
    float eav[EDC];
#pragma unroll
    for (int j = 0; j < EDC; ++j) eav[j] = ea[ed * EDC + j];
    const float* vr = v + s * DD;
    float* ar = agg + d * DD;
    for (int c = 0; c < DD; ++c) {
        float e = sbe[c];
#pragma unroll
        for (int j = 0; j < EDC; ++j) e = fmaf(eav[j], sWe[j * DD + c], e);
        atomicAdd(&ar[c], coef * (vr[c] + e));
    }
}

// combine: out = agg + skip; optional ELU; running JK max; f16 for next layer
__global__ void combine_kernel(const float* __restrict__ agg,
                               const float* __restrict__ sk,
                               float* __restrict__ jk,
                               _Float16* __restrict__ hout,
                               int doAct, int first) {
    const int i = blockIdx.x * blockDim.x + threadIdx.x;
    if (i >= NN * DD) return;
    float t = agg[i] + sk[i];
    if (doAct) t = (t > 0.f) ? t : expm1f(t);
    jk[i] = first ? t : fmaxf(jk[i], t);
    hout[i] = (_Float16)t;
}

__global__ void tanh_out_kernel(const float* __restrict__ jk,
                                float* __restrict__ out, int n) {
    const int i = blockIdx.x * blockDim.x + threadIdx.x;
    if (i < n) out[i] = tanhf(jk[i]);
}

// --------------------------------------------------------------------- host

extern "C" void kernel_launch(void* const* d_in, const int* in_sizes, int n_in,
                              void* d_out, int out_size, void* d_ws, size_t ws_size,
                              hipStream_t stream) {
    const float* x    = (const float*)d_in[0];
    const int*   eix  = (const int*)  d_in[1];
    const float* ea   = (const float*)d_in[2];
    const float* Wq0  = (const float*)d_in[3];
    const float* bq0  = (const float*)d_in[4];
    const float* Wk0  = (const float*)d_in[5];
    const float* bk0  = (const float*)d_in[6];
    const float* Wv0  = (const float*)d_in[7];
    const float* bv0  = (const float*)d_in[8];
    const float* We0  = (const float*)d_in[9];
    const float* be0  = (const float*)d_in[10];
    const float* Ws0  = (const float*)d_in[11];
    const float* bs0  = (const float*)d_in[12];
    const float* WqS  = (const float*)d_in[13];
    const float* bqS  = (const float*)d_in[14];
    const float* WkS  = (const float*)d_in[15];
    const float* bkS  = (const float*)d_in[16];
    const float* WvS  = (const float*)d_in[17];
    const float* bvS  = (const float*)d_in[18];
    const float* WeS  = (const float*)d_in[19];
    const float* beS  = (const float*)d_in[20];
    const float* WsS  = (const float*)d_in[21];
    const float* bsS  = (const float*)d_in[22];

    const int* srcI = eix;        // edge_index[0]
    const int* dstI = eix + NE;   // edge_index[1]

    // ---- workspace layout (~200 MB)
    char* ws = (char*)d_ws;
    size_t off = 0;
    auto alloc = [&](size_t bytes) -> void* {
        void* p = ws + off;
        off = (off + bytes + 255) & ~(size_t)255;
        return p;
    };
    float* q    = (float*)alloc((size_t)NN * DD * 4);
    float* k    = (float*)alloc((size_t)NN * DD * 4);
    float* v    = (float*)alloc((size_t)NN * DD * 4);
    float* sk   = (float*)alloc((size_t)NN * DD * 4);
    float* agg  = (float*)alloc((size_t)NN * DD * 4);
    float* jk   = (float*)alloc((size_t)NN * DD * 4);
    float* den  = (float*)alloc((size_t)NN * 4);
    unsigned int* amax = (unsigned int*)alloc((size_t)NN * 4);
    float* logits = (float*)alloc((size_t)NE * 4);
    _Float16* xh  = (_Float16*)alloc((size_t)NN * INC * 2);
    _Float16* hh  = (_Float16*)alloc((size_t)NN * DD * 2);
    _Float16* Wq0h = (_Float16*)alloc((size_t)INC * DD * 2);
    _Float16* Wk0h = (_Float16*)alloc((size_t)INC * DD * 2);
    _Float16* Wv0h = (_Float16*)alloc((size_t)INC * DD * 2);
    _Float16* Ws0h = (_Float16*)alloc((size_t)INC * DD * 2);
    _Float16* WqSh = (_Float16*)alloc((size_t)(NL - 1) * DD * DD * 2);
    _Float16* WkSh = (_Float16*)alloc((size_t)(NL - 1) * DD * DD * 2);
    _Float16* WvSh = (_Float16*)alloc((size_t)(NL - 1) * DD * DD * 2);
    _Float16* WsSh = (_Float16*)alloc((size_t)(NL - 1) * DD * DD * 2);

    const int TB = 256;
    auto gridFor = [](int n, int tb) { return (n + tb - 1) / tb; };

    // ---- one-time conversions to f16 (WMMA operands)
    cvt_f32_to_f16_kernel<<<gridFor(NN * INC, TB), TB, 0, stream>>>(x, xh, NN * INC);
    cvt_f32_to_f16_kernel<<<gridFor(INC * DD, TB), TB, 0, stream>>>(Wq0, Wq0h, INC * DD);
    cvt_f32_to_f16_kernel<<<gridFor(INC * DD, TB), TB, 0, stream>>>(Wk0, Wk0h, INC * DD);
    cvt_f32_to_f16_kernel<<<gridFor(INC * DD, TB), TB, 0, stream>>>(Wv0, Wv0h, INC * DD);
    cvt_f32_to_f16_kernel<<<gridFor(INC * DD, TB), TB, 0, stream>>>(Ws0, Ws0h, INC * DD);
    const int stackN = (NL - 1) * DD * DD;
    cvt_f32_to_f16_kernel<<<gridFor(stackN, TB), TB, 0, stream>>>(WqS, WqSh, stackN);
    cvt_f32_to_f16_kernel<<<gridFor(stackN, TB), TB, 0, stream>>>(WkS, WkSh, stackN);
    cvt_f32_to_f16_kernel<<<gridFor(stackN, TB), TB, 0, stream>>>(WvS, WvSh, stackN);
    cvt_f32_to_f16_kernel<<<gridFor(stackN, TB), TB, 0, stream>>>(WsS, WsSh, stackN);

    const int tiles     = (NN / 16) * (DD / 16);       // 25000 16x16 tiles
    const int gemmGrid  = gridFor(tiles, TB / 32);     // 8 waves per block
    const int edgeGrid  = NE / TB;                     // 1.6M % 256 == 0

    for (int l = 0; l < NL; ++l) {
        const _Float16* Xh = (l == 0) ? xh : hh;
        const int K = (l == 0) ? INC : DD;
        const _Float16* Wq_ = (l == 0) ? Wq0h : WqSh + (size_t)(l - 1) * DD * DD;
        const _Float16* Wk_ = (l == 0) ? Wk0h : WkSh + (size_t)(l - 1) * DD * DD;
        const _Float16* Wv_ = (l == 0) ? Wv0h : WvSh + (size_t)(l - 1) * DD * DD;
        const _Float16* Ws_ = (l == 0) ? Ws0h : WsSh + (size_t)(l - 1) * DD * DD;
        const float* bq_ = (l == 0) ? bq0 : bqS + (size_t)(l - 1) * DD;
        const float* bk_ = (l == 0) ? bk0 : bkS + (size_t)(l - 1) * DD;
        const float* bv_ = (l == 0) ? bv0 : bvS + (size_t)(l - 1) * DD;
        const float* bs_ = (l == 0) ? bs0 : bsS + (size_t)(l - 1) * DD;
        const float* We_ = (l == 0) ? We0 : WeS + (size_t)(l - 1) * EDC * DD;
        const float* be_ = (l == 0) ? be0 : beS + (size_t)(l - 1) * DD;

        gemm_qkvs_wmma_kernel<<<gemmGrid, TB, 0, stream>>>(
            Xh, Wq_, Wk_, Wv_, Ws_, bq_, bk_, bv_, bs_, q, k, v, sk, NN, K);

        zero_u32_kernel<<<gridFor(NN, TB), TB, 0, stream>>>(amax, NN);
        zero_u32_kernel<<<gridFor(NN, TB), TB, 0, stream>>>((unsigned int*)den, NN);
        zero_u32_kernel<<<gridFor(NN * DD, TB), TB, 0, stream>>>((unsigned int*)agg, NN * DD);

        edge_logits_kernel<<<edgeGrid, TB, 0, stream>>>(srcI, dstI, ea, We_, be_,
                                                        q, k, logits, amax);
        amax_finalize_kernel<<<gridFor(NN, TB), TB, 0, stream>>>(amax, NN);
        edge_exp_kernel<<<edgeGrid, TB, 0, stream>>>(dstI, logits, (const float*)amax, den);
        edge_agg_kernel<<<edgeGrid, TB, 0, stream>>>(srcI, dstI, ea, We_, be_,
                                                     v, logits, den, agg);

        combine_kernel<<<gridFor(NN * DD, TB), TB, 0, stream>>>(
            agg, sk, jk, hh, /*doAct=*/(l < NL - 1) ? 1 : 0, /*first=*/(l == 0) ? 1 : 0);
    }

    tanh_out_kernel<<<gridFor(NN * DD, TB), TB, 0, stream>>>(jk, (float*)d_out, NN * DD);
}
